// hierarchically_modular_50311246905871
// MI455X (gfx1250) — matrix operand order repaired and checked by reading, not compiled, not run on hardware
//
#include <hip/hip_runtime.h>
#include <hip/hip_bf16.h>

typedef __attribute__((ext_vector_type(16))) _Float16 v16h;
typedef __attribute__((ext_vector_type(8)))  float    v8f;

constexpr int B_   = 8192;
constexpr int D_IN = 512;
constexpr int M_   = 128;   // modules per layer (M0 == M1)
constexpr int K_   = 8;     // top-k routed inputs per module
constexpr int H_   = 128;   // module hidden width
constexpr int OUT_ = 2;

// ---------------------------------------------------------------------------
// Top-k routing: for each of Mg columns of emb[task] ([Dg, Mg] layout), find
// the k largest entries' indices (strict '>' => lowest index wins ties, same
// as jax.lax.top_k). One thread per column; Dg <= 512, k <= 8: trivial cost.
// ---------------------------------------------------------------------------
__global__ void topk_kernel(const float* __restrict__ emb,
                            const int* __restrict__ task,
                            int Dg, int Mg, int k, int* __restrict__ out_idx) {
  int m = blockIdx.x * blockDim.x + threadIdx.x;
  if (m >= Mg) return;
  const float* e = emb + (size_t)(*task) * Dg * Mg + m;
  int sel[8];
  for (int p = 0; p < k; ++p) {
    float best = -3.402823466e38f;
    int bi = 0;
    for (int d = 0; d < Dg; ++d) {
      bool taken = false;
      for (int q = 0; q < p; ++q) taken |= (sel[q] == d);
      float v = e[(size_t)d * Mg];
      if (!taken && v > best) { best = v; bi = d; }
    }
    sel[p] = bi;
    out_idx[m * k + p] = bi;
  }
}

// ---------------------------------------------------------------------------
// One modular layer: yT[m,b] = relu(gather(x,idx[m]) @ W1[m] + b1[m]).W2[m]
//                             + b2[m]
// Output is stored TRANSPOSED ([M, B]) so stores are coalesced float4s and
// the next layer's gather is coalesced across lanes.
// TIN=false: input x is [B, D] row-major (external input, scattered gather).
// TIN=true : input x is [M, B] transposed (previous layer, coalesced gather).
//
// Block = 256 threads (8 waves), owns module m = blockIdx.y and 128 batches.
// Each wave: one 16-batch tile, 8x v_wmma_f32_16x16x32_f16 over H.
//
// WMMA mapping (wave32, 16x16x32 f16, K padded 8->32):
//   A (16x32): lanes 0-15 = batch rows, elems 0-7 = gathered inputs K0..7;
//              elem 8 (K=8) of lanes 0-15 set to 1.0 so the
//              bias rides in the matrix product; all other slots zero.
//   B (32x16): lanes 0-15 = output column n, elems 0-7 = W1[m,0..7,h0+n],
//              elem 8 (K=8) = b1[m,h0+n]; lanes 16-31 are K=16..31 padding
//              and read from a zero-filled LDS mirror (no EXEC branching).
//   C = inline 0  =>  D = inp.W1 + b1 directly.
// ---------------------------------------------------------------------------
template <bool TIN>
__global__ __launch_bounds__(256)
void layer_kernel(const float* __restrict__ x, int D,
                  const int* __restrict__ idx,
                  const float* __restrict__ W1, const float* __restrict__ b1,
                  const float* __restrict__ W2, const float* __restrict__ b2,
                  float* __restrict__ y) {
  // Rows 0..H-1: B-fragment data; rows H..2H-1: zeros (for lanes 16-31).
  __shared__ alignas(32) _Float16 sWB[2 * H_][16];
  __shared__ float sW2[H_];
  __shared__ int   sIdx[K_];

  const int m   = blockIdx.y;
  const int tid = threadIdx.x;

  // Stage module parameters, pre-swizzled into the B-fragment layout.
  {
    const int r = tid;             // 256 threads -> 256 LDS rows
    if (r < H_) {
#pragma unroll
      for (int k = 0; k < K_; ++k)
        sWB[r][k] = (_Float16)W1[((size_t)m * K_ + k) * H_ + r];
      sWB[r][8] = (_Float16)b1[(size_t)m * H_ + r];
#pragma unroll
      for (int j = 9; j < 16; ++j) sWB[r][j] = (_Float16)0.f;
      sW2[r] = W2[(size_t)m * H_ + r];
    } else {
#pragma unroll
      for (int j = 0; j < 16; ++j) sWB[r][j] = (_Float16)0.f;
    }
  }
  if (tid < K_) sIdx[tid] = idx[m * K_ + tid];
  __syncthreads();

  const int wave = tid >> 5;
  const int lane = tid & 31;
  const int base = blockIdx.x * 128 + wave * 16;   // first batch row of tile
  const int col  = lane & 15;                       // N / h-within-tile index
  const int brow = (lane < 16 ? 0 : H_) + col;      // zero mirror for hi lanes

  // A fragment: one-time gather, f32 -> f16.
  v16h a = {};
  if (lane < 16) {
    const int b = base + lane;
    a[8] = (_Float16)1.f;
    if (TIN) {
#pragma unroll
      for (int k = 0; k < K_; ++k)               // coalesced across lanes
        a[k] = (_Float16)x[(size_t)sIdx[k] * B_ + b];
    } else {
      const float* xr = x + (size_t)b * D;       // scattered (L2-resident)
#pragma unroll
      for (int k = 0; k < K_; ++k) a[k] = (_Float16)xr[sIdx[k]];
    }
  }

  // Hoisted W2 chunks (one per h-tile).
  float w2s[H_ / 16];
#pragma unroll
  for (int ht = 0; ht < H_ / 16; ++ht) w2s[ht] = sW2[ht * 16 + col];

  float acc[8] = {0.f, 0.f, 0.f, 0.f, 0.f, 0.f, 0.f, 0.f};

#pragma unroll
  for (int ht = 0; ht < H_ / 16; ++ht) {
    const v16h bf = *(const v16h*)(&sWB[brow + ht * 16][0]);  // 2x ds_load_b128
    v8f c = {};                                               // inline-0 C
    c = __builtin_amdgcn_wmma_f32_16x16x32_f16(
        /*neg_a=*/false, a, /*neg_b=*/false, bf,
        /*c_mod=*/(short)0, c, /*reuse_a=*/false, /*reuse_b=*/false);
#pragma unroll
    for (int r = 0; r < 8; ++r) {
      // relu as a single v_med3: clamp to [0, FLT_MAX]
      float h = __builtin_amdgcn_fmed3f(c[r], 0.f, 3.402823466e38f);
      acc[r] += h * w2s[ht];
    }
  }

  // Reduce over the 16 N-lanes of each half (xor 1,2,4,8 stays in-half).
#pragma unroll
  for (int r = 0; r < 8; ++r) {
    float v = acc[r];
    v += __shfl_xor(v, 1, 32);
    v += __shfl_xor(v, 2, 32);
    v += __shfl_xor(v, 4, 32);
    v += __shfl_xor(v, 8, 32);
    acc[r] = v;
  }

  // C layout: VGPR r holds rows r (lanes 0-15) and r+8 (lanes 16-31).
  // Transposed store: rows are consecutive b -> two float4 stores per half.
  if (col == 0) {                       // lanes 0 and 16
    const float b2v  = b2[m];
    const int   row0 = (lane < 16) ? 0 : 8;
    float* yp = y + (size_t)m * B_ + base + row0;
    float4 s0 = make_float4(acc[0] + b2v, acc[1] + b2v, acc[2] + b2v, acc[3] + b2v);
    float4 s1 = make_float4(acc[4] + b2v, acc[5] + b2v, acc[6] + b2v, acc[7] + b2v);
    *(float4*)(yp)     = s0;
    *(float4*)(yp + 4) = s1;
  }
}

// ---------------------------------------------------------------------------
// Final head: out[b, j] = sigmoid(y1T[oidx[j], b]), j = 0..1  (coalesced)
// ---------------------------------------------------------------------------
__global__ void final_kernel(const float* __restrict__ y1T,
                             const int* __restrict__ oidx,
                             float* __restrict__ out) {
  int b = blockIdx.x * blockDim.x + threadIdx.x;
  if (b >= B_) return;
  int i0 = oidx[0], i1 = oidx[1];
  float v0 = y1T[(size_t)i0 * B_ + b];
  float v1 = y1T[(size_t)i1 * B_ + b];
  out[b * OUT_ + 0] = 1.f / (1.f + __expf(-v0));
  out[b * OUT_ + 1] = 1.f / (1.f + __expf(-v1));
}

// ---------------------------------------------------------------------------
// Launch: topk x3 -> layer0 -> layer1 -> head (all on `stream`, serialized).
// ---------------------------------------------------------------------------
extern "C" void kernel_launch(void* const* d_in, const int* in_sizes, int n_in,
                              void* d_out, int out_size, void* d_ws, size_t ws_size,
                              hipStream_t stream) {
  const float* x       = (const float*)d_in[0];
  const int*   task    = (const int*)  d_in[1];
  const float* emb0    = (const float*)d_in[2];
  const float* emb1    = (const float*)d_in[3];
  const float* emb_out = (const float*)d_in[4];
  const float* W1_0    = (const float*)d_in[5];
  const float* b1_0    = (const float*)d_in[6];
  const float* W2_0    = (const float*)d_in[7];
  const float* b2_0    = (const float*)d_in[8];
  const float* W1_1    = (const float*)d_in[9];
  const float* b1_1    = (const float*)d_in[10];
  const float* W2_1    = (const float*)d_in[11];
  const float* b2_1    = (const float*)d_in[12];

  char* ws = (char*)d_ws;
  int*   idx0 = (int*)(ws);                 // 128*8 ints   = 4 KB
  int*   idx1 = (int*)(ws + 4096);          // 128*8 ints   = 4 KB
  int*   idxo = (int*)(ws + 8192);          // 2 ints (8 KB reserved)
  float* y0   = (float*)(ws + 16384);                         // [M,B] f32 = 4 MB
  float* y1   = (float*)(ws + 16384 + (size_t)B_ * M_ * 4);   // [M,B] f32 = 4 MB

  topk_kernel<<<1, 128, 0, stream>>>(emb0,    task, D_IN, M_, K_,   idx0);
  topk_kernel<<<1, 128, 0, stream>>>(emb1,    task, M_,   M_, K_,   idx1);
  topk_kernel<<<1, 128, 0, stream>>>(emb_out, task, M_,   1,  OUT_, idxo);

  dim3 grid(B_ / 128, M_);
  layer_kernel<false><<<grid, 256, 0, stream>>>(x,  D_IN, idx0, W1_0, b1_0, W2_0, b2_0, y0);
  layer_kernel<true ><<<grid, 256, 0, stream>>>(y0, M_,   idx1, W1_1, b1_1, W2_1, b2_1, y1);

  final_kernel<<<(B_ + 255) / 256, 256, 0, stream>>>(y1, idxo, (float*)d_out);
}
